// Patient2Vec_12171937317157
// MI455X (gfx1250) — compile-verified, matrix operand order checked by reading, not run on hardware
//
#include <hip/hip_runtime.h>
#include <stdint.h>

typedef __attribute__((ext_vector_type(16))) _Float16 v16h;
typedef __attribute__((ext_vector_type(8)))  float    v8f;
typedef __attribute__((ext_vector_type(4)))  float    f4;
typedef _Float16 half_t;

#define B_   128
#define S_   48
#define P_   60
#define I_   512
#define H_   512
#define NF_  4
#define O_   2
#define G3H  (3 * H_)      // 1536
#define ROWS (B_ * S_)     // 6144
#define NT3H (G3H / 16)    // 96 N-tiles of 16

// One 16x16x32 fragment: 8 VGPRs, loadable as two b128s.
typedef union {
  v16h  v;
  uint4 q[2];
  uint32_t u[8];
} frag_t;

__device__ __forceinline__ float wave_reduce_sum(float v) {
#pragma unroll
  for (int off = 16; off > 0; off >>= 1) v += __shfl_xor(v, off, 32);
  return v;
}

__device__ __forceinline__ float sigmoidf_(float x) {
  return 1.0f / (1.0f + __expf(-x));
}

__device__ __forceinline__ uint32_t pack2h(float a, float b) {
  union { uint32_t u; half_t h[2]; } p;
  p.h[0] = (half_t)a; p.h[1] = (half_t)b;
  return p.u;
}

// ---------------------------------------------------------------------------
// Kernel 1: convert+transpose+WMMA-pack the 4 GRU weight matrices (f32 [3H,I]
// row-major) into per-lane B-fragment layout [ntile][kchunk][lane][8 dwords],
// so GEMM kernels fetch a full 16x16x32 B fragment with two b128 loads.
// B-frag layout (wave32): lane L holds column N = n0+(L&15), khalf = L>>4;
// dword v holds K = kk*32 + (v<4 ? 2v : 16+2(v-4)) + 8*khalf, and K+1.
// ---------------------------------------------------------------------------
__global__ __launch_bounds__(256) void pack_weights_kernel(
    const float* __restrict__ wih_f, const float* __restrict__ whh_f,
    const float* __restrict__ wih_b, const float* __restrict__ whh_b,
    uint32_t* __restrict__ p_wih_f, uint32_t* __restrict__ p_whh_f,
    uint32_t* __restrict__ p_wih_b, uint32_t* __restrict__ p_whh_b) {
  int gid = blockIdx.x * 256 + threadIdx.x;
  const int perMat = NT3H * 16 * 32;   // 49152 fragment-lanes per matrix
  int mat = gid / perMat;
  if (mat >= 4) return;
  int rem  = gid - mat * perMat;
  int lane = rem & 31;  rem >>= 5;
  int kk   = rem & 15;  rem >>= 4;
  int nt   = rem;                       // 0..95
  const float* src = (mat == 0) ? wih_f : (mat == 1) ? whh_f
                   : (mat == 2) ? wih_b : whh_b;
  uint32_t* dst    = (mat == 0) ? p_wih_f : (mat == 1) ? p_whh_f
                   : (mat == 2) ? p_wih_b : p_whh_b;
  const int n = nt * 16 + (lane & 15);
  const int khalf = lane >> 4;
  uint32_t u[8];
#pragma unroll
  for (int g = 0; g < 2; ++g) {
    const int kbase = kk * 32 + g * 16 + 8 * khalf;
    float4 f0 = *(const float4*)(src + (size_t)n * I_ + kbase);
    float4 f1 = *(const float4*)(src + (size_t)n * I_ + kbase + 4);
    u[g * 4 + 0] = pack2h(f0.x, f0.y);
    u[g * 4 + 1] = pack2h(f0.z, f0.w);
    u[g * 4 + 2] = pack2h(f1.x, f1.y);
    u[g * 4 + 3] = pack2h(f1.z, f1.w);
  }
  uint32_t* o = dst + (size_t)((nt * 16 + kk) * 32 + lane) * 8;
  ((uint4*)o)[0] = make_uint4(u[0], u[1], u[2], u[3]);
  ((uint4*)o)[1] = make_uint4(u[4], u[5], u[6], u[7]);
}

// ---------------------------------------------------------------------------
// Kernel 2: fused code-level attention. One WG per (b,s); single HBM pass.
// Streamed 755MB input uses NON-TEMPORAL loads (don't wash the 192MB L2 that
// downstream GEMM weights/xp want to live in). alpha is write-once -> NT store.
// ---------------------------------------------------------------------------
__global__ __launch_bounds__(256) void attn_pool_kernel(
    const float* __restrict__ inputs, const float* __restrict__ conv_w,
    const float* __restrict__ conv_b, float* __restrict__ alpha_out,
    half_t* __restrict__ emb16) {
  const int bs = blockIdx.x;                      // 0..6143
  const int tid = threadIdx.x, lane = tid & 31, wave = tid >> 5;
  const float* base = inputs + (size_t)bs * P_ * I_;
  float cw[16];
  {
    const f4* c4 = (const f4*)(conv_w + lane * 16);
#pragma unroll
    for (int q = 0; q < 4; ++q) {
      f4 t = c4[q];
      cw[4 * q + 0] = t.x; cw[4 * q + 1] = t.y;
      cw[4 * q + 2] = t.z; cw[4 * q + 3] = t.w;
    }
  }
  const float cb = conv_b[0];
  float acc[16];
#pragma unroll
  for (int i = 0; i < 16; ++i) acc[i] = 0.0f;

  for (int p = wave; p < P_; p += 8) {
    if (p + 8 < P_)
      __builtin_prefetch(base + (size_t)(p + 8) * I_ + lane * 16, 0, 1);
    const f4* x4 = (const f4*)(base + (size_t)p * I_ + lane * 16);
    float x[16];
#pragma unroll
    for (int q = 0; q < 4; ++q) {
      f4 t = __builtin_nontemporal_load(x4 + q);   // TH=NT stream
      x[4 * q + 0] = t.x; x[4 * q + 1] = t.y;
      x[4 * q + 2] = t.z; x[4 * q + 3] = t.w;
    }
    float d = 0.0f;
#pragma unroll
    for (int i = 0; i < 16; ++i) d += x[i] * cw[i];
    d = wave_reduce_sum(d);
    float a = d + cb;
    a = fminf(fmaxf(a, 0.0f), 1.0f);               // hardtanh(0,1)
    if (lane == 0)
      __builtin_nontemporal_store(a, alpha_out + (size_t)bs * P_ + p);
#pragma unroll
    for (int i = 0; i < 16; ++i) acc[i] += a * x[i];
  }

  __shared__ float red[8][I_];
#pragma unroll
  for (int i = 0; i < 16; ++i) red[wave][lane * 16 + i] = acc[i];
  __syncthreads();
#pragma unroll
  for (int r = 0; r < 2; ++r) {
    const int i = tid * 2 + r;                     // 0..511
    float s = 0.0f;
#pragma unroll
    for (int w = 0; w < 8; ++w) s += red[w][i];
    emb16[(size_t)bs * I_ + i] = (half_t)s;
  }
}

// ---------------------------------------------------------------------------
// Kernel 3: xp = emb @ Wih^T + bih, both directions, WMMA f16->f32.
// Each wave: 1 Mtile x 4 Ntiles. Base pointers hoisted; all fragments of a
// k-iteration loaded before the WMMA sequence (immediate-offset b128 clause,
// partial loadcnt waits overlap later loads with earlier WMMAs).
// ---------------------------------------------------------------------------
__global__ __launch_bounds__(256) void xp_gemm_kernel(
    const half_t* __restrict__ emb16,
    const uint32_t* __restrict__ p_wih_f, const uint32_t* __restrict__ p_wih_b,
    const float* __restrict__ bih_f, const float* __restrict__ bih_b,
    float* __restrict__ xp_f, float* __restrict__ xp_b) {
  const int lane = threadIdx.x & 31, wave = threadIdx.x >> 5;
  const int mt  = blockIdx.x;                         // 0..383
  const int nt0 = (blockIdx.y * 8 + wave) * 4;        // 0,4,...,92
  const int dir = blockIdx.z;
  const uint32_t* Bp = dir ? p_wih_b : p_wih_f;
  const float* bias  = dir ? bih_b   : bih_f;
  float* xp          = dir ? xp_b    : xp_f;
  const int m0 = mt * 16;
  const int khalf = lane >> 4;

  v8f acc[4];
#pragma unroll
  for (int t = 0; t < 4; ++t)
#pragma unroll
    for (int i = 0; i < 8; ++i) acc[t][i] = 0.0f;

  // Hoisted per-wave bases; loop indices become immediate offsets.
  const half_t* aw = emb16 + (size_t)(m0 + (lane & 15)) * I_ + 8 * khalf;
  const uint4*  bw = (const uint4*)(Bp + (size_t)nt0 * 16 * 256 + (size_t)lane * 8);

  for (int kk = 0; kk < 16; ++kk) {
    frag_t A, Bf[4];
    const uint4* ap = (const uint4*)(aw + kk * 32);
    const uint4* bp = bw + kk * 64;                   // 256 dwords per kchunk
    A.q[0] = ap[0];
    A.q[1] = ap[2];                                   // +16 halves
#pragma unroll
    for (int t = 0; t < 4; ++t) {                     // tile stride: 1024 uint4s
      Bf[t].q[0] = bp[t * 1024];
      Bf[t].q[1] = bp[t * 1024 + 1];
    }
#pragma unroll
    for (int t = 0; t < 4; ++t)
      acc[t] = __builtin_amdgcn_wmma_f32_16x16x32_f16(false, A.v, false, Bf[t].v,
                                                      (short)0, acc[t], false, false);
  }
#pragma unroll
  for (int t = 0; t < 4; ++t) {
    const int n = (nt0 + t) * 16 + (lane & 15);
    const float bn = bias[n];
#pragma unroll
    for (int r = 0; r < 8; ++r) {
      const int m = m0 + r + 8 * khalf;              // C/D layout: VGPR r, hi-half -> M+8
      xp[(size_t)m * G3H + n] = acc[t][r] + bn;
    }
  }
}

// ---------------------------------------------------------------------------
// Kernel 4: GRU recurrence. 16 chains (8 batch-tiles x 2 dirs), one WG each,
// 1024 threads (32 waves, 8/SIMD); each wave owns 3 N-tiles. Fragment bases
// hoisted out of the step loop; per k-iteration all 2 A (ds b128) + 6 B
// (global b128) loads issue before the 3-WMMA sequence.
// LDS (dynamic, 144KB): h f32 [16][512] + h f16 [16][512] + hp f32 [16][1536].
// ---------------------------------------------------------------------------
__global__ __launch_bounds__(1024) void gru_chain_kernel(
    const uint32_t* __restrict__ p_whh_f, const uint32_t* __restrict__ p_whh_b,
    const float* __restrict__ bhh_f, const float* __restrict__ bhh_b,
    const float* __restrict__ xp_f, const float* __restrict__ xp_b,
    float* __restrict__ states) {
  extern __shared__ char smem[];
  float*  h32 = (float*)smem;                               // 32 KB
  half_t* h16 = (half_t*)(smem + 16 * H_ * 4);              // 16 KB
  float*  hp  = (float*)(smem + 16 * H_ * 4 + 16 * H_ * 2); // 96 KB

  const int chain = blockIdx.x;                  // 0..15
  const int dir = chain >> 3, mt = chain & 7;
  const int m0 = mt * 16;
  const uint32_t* Bp = dir ? p_whh_b : p_whh_f;
  const float* bhh   = dir ? bhh_b   : bhh_f;
  const float* xp    = dir ? xp_b    : xp_f;
  const int tid = threadIdx.x, lane = tid & 31, wave = tid >> 5;
  const int khalf = lane >> 4;
  const int nt0 = wave * 3;                      // 3 N-tiles per wave (96/32)

  for (int idx = tid; idx < 16 * H_; idx += 1024) {
    h32[idx] = 0.0f;
    h16[idx] = (half_t)0.0f;
  }
  __syncthreads();

  // Loop-invariant fragment base pointers.
  const half_t* aw = h16 + (size_t)(lane & 15) * H_ + 8 * khalf;
  const uint4*  bw = (const uint4*)(Bp + (size_t)nt0 * 16 * 256 + (size_t)lane * 8);
  float* hpw = hp + 8 * khalf * G3H + (lane & 15);

  for (int step = 0; step < S_; ++step) {
    const int t = dir ? (S_ - 1 - step) : step;

    // --- hp = h @ Whh^T : 3 tiles/wave ---
    v8f acc[3];
#pragma unroll
    for (int tt = 0; tt < 3; ++tt)
#pragma unroll
      for (int i = 0; i < 8; ++i) acc[tt][i] = 0.0f;

    for (int kk = 0; kk < 16; ++kk) {
      frag_t A, Bf[3];
      const uint4* ap = (const uint4*)(aw + kk * 32);
      const uint4* bp = bw + kk * 64;
      A.q[0] = ap[0];                               // ds_load_b128
      A.q[1] = ap[2];                               // ds_load_b128 (+16 halves)
#pragma unroll
      for (int tt = 0; tt < 3; ++tt) {              // tile stride: 1024 uint4s
        Bf[tt].q[0] = bp[tt * 1024];
        Bf[tt].q[1] = bp[tt * 1024 + 1];
      }
#pragma unroll
      for (int tt = 0; tt < 3; ++tt)
        acc[tt] = __builtin_amdgcn_wmma_f32_16x16x32_f16(false, A.v, false, Bf[tt].v,
                                                         (short)0, acc[tt], false, false);
    }
#pragma unroll
    for (int tt = 0; tt < 3; ++tt) {
#pragma unroll
      for (int r = 0; r < 8; ++r)
        hpw[r * G3H + (nt0 + tt) * 16] = acc[tt][r];
    }
    __syncthreads();

    // --- gates: r,z,n (PyTorch order), h_new = (1-z)*n + z*h ---
    for (int idx = tid; idx < 16 * H_; idx += 1024) {
      const int m = idx >> 9, j = idx & (H_ - 1);
      const size_t row = (size_t)(m0 + m) * S_ + t;
      const float* xr  = xp + row * G3H;
      const float* hpr = hp + (size_t)m * G3H;
      float rg = sigmoidf_(xr[j]      + hpr[j]      + bhh[j]);
      float zg = sigmoidf_(xr[H_ + j] + hpr[H_ + j] + bhh[H_ + j]);
      float ng = tanhf(xr[2 * H_ + j] + rg * (hpr[2 * H_ + j] + bhh[2 * H_ + j]));
      float hnew = (1.0f - zg) * ng + zg * h32[idx];
      h32[idx] = hnew;
      h16[idx] = (half_t)hnew;
      states[row * (2 * H_) + dir * H_ + j] = hnew;
    }
    __syncthreads();
  }
}

// ---------------------------------------------------------------------------
// Kernel 5: beta attention + final linear + softmaxes. One WG per batch row.
// ---------------------------------------------------------------------------
__global__ __launch_bounds__(256) void attn_out_kernel(
    const float* __restrict__ states, const float* __restrict__ conv2_w,
    const float* __restrict__ conv2_b, const float* __restrict__ lin_w,
    const float* __restrict__ lin_b, float* __restrict__ out,
    float* __restrict__ beta_out) {
  const int b = blockIdx.x;
  const int tid = threadIdx.x, lane = tid & 31, wave = tid >> 5;
  __shared__ float logits[NF_][S_];
  __shared__ float beta[NF_][S_];
  __shared__ float ctx[NF_ * 2 * H_];
  __shared__ float red2[2][8];
  const float* st = states + (size_t)b * S_ * (2 * H_);

  // logits[f][s] = states[b,s,:] . conv2_w[f,:] + conv2_b[f]
  for (int w = wave; w < NF_ * S_; w += 8) {
    const int f = w / S_, s = w % S_;
    const float* sv = st + (size_t)s * (2 * H_);
    const float* cw = conv2_w + (size_t)f * (2 * H_);
    float d = 0.0f;
    for (int h = lane; h < 2 * H_; h += 32) d += sv[h] * cw[h];
    d = wave_reduce_sum(d);
    if (lane == 0) logits[f][s] = d + conv2_b[f];
  }
  __syncthreads();

  // softmax over s per filter; emit beta
  if (tid < NF_) {
    const int f = tid;
    float mx = -1e30f;
    for (int s = 0; s < S_; ++s) mx = fmaxf(mx, logits[f][s]);
    float sum = 0.0f;
    for (int s = 0; s < S_; ++s) { float e = __expf(logits[f][s] - mx); beta[f][s] = e; sum += e; }
    const float inv = 1.0f / sum;
    for (int s = 0; s < S_; ++s) {
      beta[f][s] *= inv;
      beta_out[((size_t)b * NF_ + f) * S_ + s] = beta[f][s];
    }
  }
  __syncthreads();

  // context[f][h] = sum_s beta[f][s] * states[b,s,h]
  for (int idx = tid; idx < NF_ * 2 * H_; idx += 256) {
    const int f = idx / (2 * H_), h = idx & (2 * H_ - 1);
    float s = 0.0f;
    for (int t = 0; t < S_; ++t) s += beta[f][t] * st[(size_t)t * (2 * H_) + h];
    ctx[idx] = s;
  }
  __syncthreads();

  // out = softmax(ctx_flat @ lin_w^T + lin_b), O = 2
  float p0 = 0.0f, p1 = 0.0f;
  for (int idx = tid; idx < NF_ * 2 * H_; idx += 256) {
    const float c = ctx[idx];
    p0 += c * lin_w[idx];
    p1 += c * lin_w[NF_ * 2 * H_ + idx];
  }
  p0 = wave_reduce_sum(p0);
  p1 = wave_reduce_sum(p1);
  if (lane == 0) { red2[0][wave] = p0; red2[1][wave] = p1; }
  __syncthreads();
  if (tid == 0) {
    float a0 = lin_b[0], a1 = lin_b[1];
    for (int w = 0; w < 8; ++w) { a0 += red2[0][w]; a1 += red2[1][w]; }
    const float mx = fmaxf(a0, a1);
    const float e0 = __expf(a0 - mx), e1 = __expf(a1 - mx);
    const float inv = 1.0f / (e0 + e1);
    out[b * 2 + 0] = e0 * inv;
    out[b * 2 + 1] = e1 * inv;
  }
}

// ---------------------------------------------------------------------------
extern "C" void kernel_launch(void* const* d_in, const int* in_sizes, int n_in,
                              void* d_out, int out_size, void* d_ws, size_t ws_size,
                              hipStream_t stream) {
  (void)in_sizes; (void)n_in; (void)out_size; (void)ws_size;
  const float* inputs  = (const float*)d_in[0];
  const float* conv_w  = (const float*)d_in[1];
  const float* conv_b  = (const float*)d_in[2];
  const float* wih_f   = (const float*)d_in[3];
  const float* whh_f   = (const float*)d_in[4];
  const float* bih_f   = (const float*)d_in[5];
  const float* bhh_f   = (const float*)d_in[6];
  const float* wih_b   = (const float*)d_in[7];
  const float* whh_b   = (const float*)d_in[8];
  const float* bih_b   = (const float*)d_in[9];
  const float* bhh_b   = (const float*)d_in[10];
  const float* conv2_w = (const float*)d_in[11];
  const float* conv2_b = (const float*)d_in[12];
  const float* lin_w   = (const float*)d_in[13];
  const float* lin_b   = (const float*)d_in[14];

  float* out_p   = (float*)d_out;                  // [128,2]
  float* alpha_p = out_p + B_ * O_;                // [128,48,60]
  float* beta_p  = alpha_p + (size_t)B_ * S_ * P_; // [128,4,48]

  char* ws = (char*)d_ws;
  size_t off = 0;
  auto wsalloc = [&](size_t bytes) -> void* {
    void* p = ws + off;
    off = (off + bytes + 255) & ~(size_t)255;
    return p;
  };
  half_t*   emb16   = (half_t*)  wsalloc((size_t)ROWS * I_ * 2);
  float*    xp_f    = (float*)   wsalloc((size_t)ROWS * G3H * 4);
  float*    xp_b    = (float*)   wsalloc((size_t)ROWS * G3H * 4);
  float*    states  = (float*)   wsalloc((size_t)B_ * S_ * 2 * H_ * 4);
  uint32_t* p_wih_f = (uint32_t*)wsalloc((size_t)G3H * I_ * 2);
  uint32_t* p_whh_f = (uint32_t*)wsalloc((size_t)G3H * H_ * 2);
  uint32_t* p_wih_b = (uint32_t*)wsalloc((size_t)G3H * I_ * 2);
  uint32_t* p_whh_b = (uint32_t*)wsalloc((size_t)G3H * H_ * 2);

  const int gruLds = 16 * H_ * 4 + 16 * H_ * 2 + 16 * G3H * 4;  // 147456 B
  (void)hipFuncSetAttribute((const void*)gru_chain_kernel,
                            hipFuncAttributeMaxDynamicSharedMemorySize, gruLds);

  pack_weights_kernel<<<768, 256, 0, stream>>>(wih_f, whh_f, wih_b, whh_b,
                                               p_wih_f, p_whh_f, p_wih_b, p_whh_b);
  attn_pool_kernel<<<ROWS, 256, 0, stream>>>(inputs, conv_w, conv_b,
                                             alpha_p, emb16);
  xp_gemm_kernel<<<dim3(ROWS / 16, 3, 2), 256, 0, stream>>>(
      emb16, p_wih_f, p_wih_b, bih_f, bih_b, xp_f, xp_b);
  gru_chain_kernel<<<16, 1024, gruLds, stream>>>(p_whh_f, p_whh_b, bhh_f, bhh_b,
                                                 xp_f, xp_b, states);
  attn_out_kernel<<<B_, 256, 0, stream>>>(states, conv2_w, conv2_b,
                                          lin_w, lin_b, out_p, beta_p);
}
